// Decoder_F_12120397709391
// MI455X (gfx1250) — matrix operand order repaired and checked by reading, not compiled, not run on hardware
//
#include <hip/hip_runtime.h>
#include <stdint.h>

// ---------------------------------------------------------------------------
// Types for CDNA5 WMMA (wave32, 16x16x32 bf16 -> f32)
// ---------------------------------------------------------------------------
typedef __attribute__((ext_vector_type(16))) __bf16 v16bf;
typedef __attribute__((ext_vector_type(8)))  float  v8f;

#define BM 128          // block tile M
#define BN 64           // block tile N
#define BK 64           // K per staging round (2 WMMA k-slices)
#define LDSP 72         // padded LDS row stride in ushorts (144B, 16B aligned)
#define KDIM 512        // compile-time K for every GEMM in this model

__device__ __forceinline__ unsigned short f2bf(float f) {
  unsigned int u = __float_as_uint(f);
  u += 0x7FFFu + ((u >> 16) & 1u);           // round-to-nearest-even
  return (unsigned short)(u >> 16);
}

// ---------------------------------------------------------------------------
// fp32 -> bf16 conversion, 4 elements / thread (for emb)
// ---------------------------------------------------------------------------
__global__ void cvt_bf16x4_kernel(const float4* __restrict__ src,
                                  ushort4* __restrict__ dst, int n4) {
  int i = blockIdx.x * blockDim.x + threadIdx.x;
  if (i < n4) {
    float4 f = src[i];
    ushort4 o;
    o.x = f2bf(f.x); o.y = f2bf(f.y); o.z = f2bf(f.z); o.w = f2bf(f.w);
    dst[i] = o;
  }
}

// ---------------------------------------------------------------------------
// Weight convert + transpose: W[K][N] f32 -> Wt[N][K] bf16 (done once; tiny)
// ---------------------------------------------------------------------------
__global__ __launch_bounds__(256)
void cvt_transpose_bf16_kernel(const float* __restrict__ src,
                               unsigned short* __restrict__ dst,
                               int K, int N) {
  __shared__ float tile[32][33];
  const int n0 = blockIdx.x * 32;
  const int k0 = blockIdx.y * 32;
  const int tx = threadIdx.x & 31;
  const int ty = threadIdx.x >> 5;          // 0..7
#pragma unroll
  for (int i = ty; i < 32; i += 8)
    tile[i][tx] = src[(size_t)(k0 + i) * N + n0 + tx];
  __syncthreads();
#pragma unroll
  for (int i = ty; i < 32; i += 8)
    dst[(size_t)(n0 + i) * K + k0 + tx] = f2bf(tile[tx][i]);
}

// ---------------------------------------------------------------------------
// bf16 GEMM: out[M][N] = op(A[M][512] * Bt[N][512]^T + bias [+ addsrc])
// 256 threads = 8 waves (4x2); wave tile 32x32 = 2x2 WMMA tiles; BK=64 ->
// 8 v_wmma per staged tile. Tile i+1's six global_load_b128 (single base
// VGPR + immediate offsets) are issued right after tile i is published to
// LDS; only the next iteration's ds_store waits on them.
// ---------------------------------------------------------------------------
template <bool RELU, bool BF16OUT, int N>
__global__ __launch_bounds__(256)
void gemm_bf16_kernel(const unsigned short* __restrict__ A,   // [M][512] bf16
                      const unsigned short* __restrict__ Bt,  // [N][512] bf16
                      const float* __restrict__ bias,         // [N]
                      const float* __restrict__ addsrc,       // [M][N] or null
                      void* __restrict__ out) {
  constexpr int K = KDIM;
  constexpr size_t ASTEP = (size_t)32 * K;        // elems between staged rows
  constexpr int    SSTEP = 32 * LDSP;             // LDS elems between rows

  __shared__ __align__(16) unsigned short As[BM][LDSP];
  __shared__ __align__(16) unsigned short Bs[BN][LDSP];

  const int tid  = threadIdx.x;
  const int lane = tid & 31;
  const int wave = tid >> 5;
  const int wm   = wave >> 1;        // 0..3
  const int wn   = wave & 1;         // 0..1
  const int row0 = blockIdx.y * BM;
  const int col0 = blockIdx.x * BN;
  const int lr   = lane & 15;
  const int hs   = lane >> 4;        // half-wave select
  const int kbA  = hs ? 8 : 0;       // A: lanes 16-31 hold K=8..15 / 24..31
  const int kbB  = hs ? 8 : 0;       // B: lanes 16-31 hold K=8..15 / 24..31

  // single base address per matrix; everything else is immediate offsets
  const int tr = tid >> 3;           // 0..31
  const int tc = (tid & 7) << 3;     // 0..56 step 8 ushorts
  const unsigned short* ga = A  + (size_t)(row0 + tr) * K + tc;
  const unsigned short* gb = Bt + (size_t)(col0 + tr) * K + tc;
  unsigned short* sa = &As[tr][tc];
  unsigned short* sb = &Bs[tr][tc];

  v8f c[2][2] = {};
  uint4 a0, a1, a2, a3, b0, b1;

  // prologue: fetch first tile into registers
  a0 = *(const uint4*)(ga + 0 * ASTEP);
  a1 = *(const uint4*)(ga + 1 * ASTEP);
  a2 = *(const uint4*)(ga + 2 * ASTEP);
  a3 = *(const uint4*)(ga + 3 * ASTEP);
  b0 = *(const uint4*)(gb + 0 * ASTEP);
  b1 = *(const uint4*)(gb + 1 * ASTEP);
  ga += BK; gb += BK;

  for (int k0 = 0; k0 < K; k0 += BK) {
    __syncthreads();                 // previous tile's fragment reads done
    *(uint4*)(sa + 0 * SSTEP) = a0;
    *(uint4*)(sa + 1 * SSTEP) = a1;
    *(uint4*)(sa + 2 * SSTEP) = a2;
    *(uint4*)(sa + 3 * SSTEP) = a3;
    *(uint4*)(sb + 0 * SSTEP) = b0;
    *(uint4*)(sb + 1 * SSTEP) = b1;
    __syncthreads();                 // tile published

    if (k0 + BK < K) {               // issue next tile's loads; don't wait
      a0 = *(const uint4*)(ga + 0 * ASTEP);
      a1 = *(const uint4*)(ga + 1 * ASTEP);
      a2 = *(const uint4*)(ga + 2 * ASTEP);
      a3 = *(const uint4*)(ga + 3 * ASTEP);
      b0 = *(const uint4*)(gb + 0 * ASTEP);
      b1 = *(const uint4*)(gb + 1 * ASTEP);
      ga += BK; gb += BK;
    }

    // ---- two WMMA k-slices per staged tile ----
    union FragU { v16bf v; uint4 q[2]; };
#pragma unroll
    for (int ks = 0; ks < 2; ++ks) {
      const int kof = ks * 32;
      FragU a[2], b[2];
#pragma unroll
      for (int i = 0; i < 2; ++i) {
        int m = wm * 32 + i * 16 + lr;
        a[i].q[0] = *(const uint4*)&As[m][kof + kbA];
        a[i].q[1] = *(const uint4*)&As[m][kof + kbA + 16];
      }
#pragma unroll
      for (int j = 0; j < 2; ++j) {
        int n = wn * 32 + j * 16 + lr;
        b[j].q[0] = *(const uint4*)&Bs[n][kof + kbB];
        b[j].q[1] = *(const uint4*)&Bs[n][kof + kbB + 16];
      }
#pragma unroll
      for (int i = 0; i < 2; ++i)
#pragma unroll
        for (int j = 0; j < 2; ++j)
          c[i][j] = __builtin_amdgcn_wmma_f32_16x16x32_bf16(
              false, a[i].v, false, b[j].v, (short)0, c[i][j], false, false);
    }
  }

  // ---- epilogue: bias (+ReLU) (+residual add), fused store ----
#pragma unroll
  for (int i = 0; i < 2; ++i) {
#pragma unroll
    for (int j = 0; j < 2; ++j) {
#pragma unroll
      for (int r = 0; r < 8; ++r) {
        int m = row0 + wm * 32 + i * 16 + r + hs * 8;   // C/D VGPR layout
        int n = col0 + wn * 32 + j * 16 + lr;
        float v = c[i][j][r] + bias[n];
        if (RELU) v = fmaxf(v, 0.0f);
        if (addsrc) v += addsrc[(size_t)m * N + n];
        if (BF16OUT)
          ((unsigned short*)out)[(size_t)m * N + n] = f2bf(v);
        else
          ((float*)out)[(size_t)m * N + n] = v;
      }
    }
  }
}

// ---------------------------------------------------------------------------
// kNN (k=3) inverse-distance interpolation: one wave32 per query point.
// ---------------------------------------------------------------------------
__device__ __forceinline__ void top3_insert(float nd, int ni,
                                            float& d0, float& d1, float& d2,
                                            int& i0, int& i1, int& i2) {
  if (nd < d2) {
    if (nd < d1) {
      d2 = d1; i2 = i1;
      if (nd < d0) { d1 = d0; i1 = i0; d0 = nd; i0 = ni; }
      else         { d1 = nd; i1 = ni; }
    } else { d2 = nd; i2 = ni; }
  }
}

__global__ __launch_bounds__(256)
void knn3_kernel(const float* __restrict__ l_y,    // [NL][F]
                 const float* __restrict__ l_pos,  // [NL][3]
                 const float* __restrict__ h_pos,  // [NH][3]
                 float* __restrict__ out,          // [NH][F]
                 int NL, int NH, int F) {
  int q    = blockIdx.x * (blockDim.x >> 5) + (threadIdx.x >> 5);
  int lane = threadIdx.x & 31;
  if (q >= NH) return;

  float hx = h_pos[q * 3 + 0];
  float hy = h_pos[q * 3 + 1];
  float hz = h_pos[q * 3 + 2];

  float d0 = 3.4e38f, d1 = 3.4e38f, d2 = 3.4e38f;
  int   i0 = 0, i1 = 0, i2 = 0;

  for (int j = lane; j < NL; j += 32) {
    float dx = hx - l_pos[j * 3 + 0];
    float dy = hy - l_pos[j * 3 + 1];
    float dz = hz - l_pos[j * 3 + 2];
    float dd = fmaf(dx, dx, fmaf(dy, dy, dz * dz));
    top3_insert(dd, j, d0, d1, d2, i0, i1, i2);
  }
  // wave32 butterfly merge of per-lane top-3 lists
  for (int off = 16; off >= 1; off >>= 1) {
    float od0 = __shfl_down(d0, off), od1 = __shfl_down(d1, off), od2 = __shfl_down(d2, off);
    int   oi0 = __shfl_down(i0, off), oi1 = __shfl_down(i1, off), oi2 = __shfl_down(i2, off);
    top3_insert(od0, oi0, d0, d1, d2, i0, i1, i2);
    top3_insert(od1, oi1, d0, d1, d2, i0, i1, i2);
    top3_insert(od2, oi2, d0, d1, d2, i0, i1, i2);
  }
  d0 = __shfl(d0, 0); d1 = __shfl(d1, 0); d2 = __shfl(d2, 0);
  i0 = __shfl(i0, 0); i1 = __shfl(i1, 0); i2 = __shfl(i2, 0);

  float w0 = 1.0f / fmaxf(d0, 1e-16f);
  float w1 = 1.0f / fmaxf(d1, 1e-16f);
  float w2 = 1.0f / fmaxf(d2, 1e-16f);
  float inv_ws = 1.0f / (w0 + w1 + w2);

  for (int f = lane; f < F; f += 32) {       // coalesced across lanes
    float v = w0 * l_y[(size_t)i0 * F + f] +
              w1 * l_y[(size_t)i1 * F + f] +
              w2 * l_y[(size_t)i2 * F + f];
    out[(size_t)q * F + f] = v * inv_ws;
  }
}

// ---------------------------------------------------------------------------
// Launch: cvt/transpose -> GEMM1(relu) -> GEMM2(relu) -> kNN -> GEMM3(+knn)
// Workspace layout (reused):
//   [0, 32M)   embh (bf16)  -> later x2 (bf16)
//   [32M,64M)  x1   (bf16)  -> later knn result (f32, 16MB)
//   [64M,...)  W1t, W2t, W3t (bf16, transposed [N][K])
// ---------------------------------------------------------------------------
extern "C" void kernel_launch(void* const* d_in, const int* in_sizes, int n_in,
                              void* d_out, int out_size, void* d_ws, size_t ws_size,
                              hipStream_t stream) {
  const float* emb   = (const float*)d_in[0];
  const float* l_y   = (const float*)d_in[1];
  const float* l_pos = (const float*)d_in[2];
  const float* h_pos = (const float*)d_in[3];
  const float* W1    = (const float*)d_in[4];
  const float* b1    = (const float*)d_in[5];
  const float* W2    = (const float*)d_in[6];
  const float* b2    = (const float*)d_in[7];
  const float* W3    = (const float*)d_in[8];
  const float* b3    = (const float*)d_in[9];

  const int NH = 32768, NL = 8192, H = 512, O = 128;

  char* ws = (char*)d_ws;
  unsigned short* embh = (unsigned short*)(ws);                       // NH*H bf16
  unsigned short* x1   = (unsigned short*)(ws + (size_t)NH * H * 2);  // NH*H bf16
  unsigned short* w1t  = (unsigned short*)(ws + (size_t)NH * H * 4);
  unsigned short* w2t  = w1t + (size_t)H * H;
  unsigned short* w3t  = w2t + (size_t)H * H;
  unsigned short* x2   = embh;          // reuse after GEMM1 consumed embh
  float*          knn  = (float*)x1;    // reuse after GEMM2 consumed x1

  // --- emb -> bf16 ---
  {
    int n4 = (NH * H) / 4;
    cvt_bf16x4_kernel<<<n4 / 256, 256, 0, stream>>>((const float4*)emb, (ushort4*)embh, n4);
  }
  // --- weights -> bf16, transposed to [N][K] ---
  cvt_transpose_bf16_kernel<<<dim3(H / 32, H / 32), 256, 0, stream>>>(W1, w1t, H, H);
  cvt_transpose_bf16_kernel<<<dim3(H / 32, H / 32), 256, 0, stream>>>(W2, w2t, H, H);
  cvt_transpose_bf16_kernel<<<dim3(O / 32, H / 32), 256, 0, stream>>>(W3, w3t, H, O);

  // --- GEMM1: x1 = relu(emb @ W1 + b1)  [bf16 out] ---
  {
    dim3 grid(512 / BN, NH / BM);
    gemm_bf16_kernel<true, true, 512><<<grid, 256, 0, stream>>>(
        embh, w1t, b1, nullptr, (void*)x1);
  }
  // --- GEMM2: x2 = relu(x1 @ W2 + b2)  [bf16 out, overwrites embh] ---
  {
    dim3 grid(512 / BN, NH / BM);
    gemm_bf16_kernel<true, true, 512><<<grid, 256, 0, stream>>>(
        x1, w2t, b2, nullptr, (void*)x2);
  }
  // --- kNN interpolate into knn buffer (overwrites x1 region) ---
  {
    int waves_per_block = 256 / 32;
    knn3_kernel<<<NH / waves_per_block, 256, 0, stream>>>(
        l_y, l_pos, h_pos, knn, NL, NH, O);
  }
  // --- GEMM3: d_out = x2 @ W3 + b3 + knn  [f32 out] ---
  {
    dim3 grid(128 / BN, NH / BM);
    gemm_bf16_kernel<false, false, 128><<<grid, 256, 0, stream>>>(
        x2, w3t, b3, knn, d_out);
  }
}
